// WindowAttention_30923764531331
// MI455X (gfx1250) — matrix operand order repaired and checked by reading, not compiled, not run on hardware
//
#include <hip/hip_runtime.h>

// ---------------------------------------------------------------------------
// MI455X (gfx1250) window-attention pipeline, bf16 WMMA everywhere.
// Shapes: B=128, N=512, C=256, H=8, HD=32, NW=16.
// ---------------------------------------------------------------------------

typedef __bf16 bf16;
typedef bf16  v16bf __attribute__((ext_vector_type(16)));
typedef bf16  v8bf  __attribute__((ext_vector_type(8)));
typedef float v8f   __attribute__((ext_vector_type(8)));

#define BATCH 128
#define NTOK  512
#define DIM   256
#define HEADS 8
#define HD    32
#define NW    16

// ---------------- workspace layout (bytes) ----------------
// table  : 3375*8 f32
// biasH  : 8*512*512 f32      (16*sigmoid(cpb))
// xb     : B*N*C bf16
// qkvw   : 768*256 bf16
// projw  : 256*256 bf16
// q,k    : (B*H, N, HD) bf16
// vT     : (B*H, HD, N) bf16  (transposed for contiguous B-fragments)
// ob     : B*N*C bf16         (attention output, proj input)
static constexpr size_t OFF_TABLE = 0;
static constexpr size_t OFF_BIAS  = 131072;                         // 128 KiB pad
static constexpr size_t OFF_XB    = OFF_BIAS  + (size_t)8*512*512*4;
static constexpr size_t OFF_QKVW  = OFF_XB    + (size_t)BATCH*NTOK*DIM*2;
static constexpr size_t OFF_PROJW = OFF_QKVW  + (size_t)768*256*2;
static constexpr size_t OFF_Q     = OFF_PROJW + (size_t)256*256*2;
static constexpr size_t OFF_K     = OFF_Q     + (size_t)BATCH*HEADS*NTOK*HD*2;
static constexpr size_t OFF_VT    = OFF_K     + (size_t)BATCH*HEADS*NTOK*HD*2;
static constexpr size_t OFF_OB    = OFF_VT    + (size_t)BATCH*HEADS*NTOK*HD*2;

// ---------------- WMMA fragment helpers ----------------
union V16U { v16bf v; v8bf h[2]; };

// A-matrix 16x32 bf16: lane(0-15) row=lane, K in {0..15};
// lane(16-31) row=lane-16, K in {16..31}. `row` points at K=0 of this tile.
__device__ __forceinline__ v16bf load_a_frag(const bf16* row, int half) {
    V16U u;
    u.h[0] = *reinterpret_cast<const v8bf*>(row + 16 * half);
    u.h[1] = *reinterpret_cast<const v8bf*>(row + 16 * half + 8);
    return u.v;
}

// B-matrix 32x16 bf16: lane l holds column l%16, K range [16*(l>>4), +16) contiguous.
// `colrow` points at K=0 for this lane's column.
__device__ __forceinline__ v16bf load_b_frag(const bf16* colrow, int half) {
    return *reinterpret_cast<const v16bf*>(colrow + 16 * half);
}

__device__ __forceinline__ v8f wmma_bf16(v16bf a, v16bf b, v8f c) {
    return __builtin_amdgcn_wmma_f32_16x16x32_bf16(false, a, false, b, (short)0, c,
                                                   false, false);
}

// ---------------- prep kernels ----------------
__global__ void cast_f32_to_bf16(const float* __restrict__ src,
                                 bf16* __restrict__ dst, int n) {
    int i = blockIdx.x * blockDim.x + threadIdx.x;
    if (i < n) dst[i] = (bf16)src[i];
}

// CPB MLP: table[r][h] = cpb_w2[h] . relu(cpb_w1 @ rpb[r] + cpb_b1)
__global__ void cpb_table_kernel(const float* __restrict__ rpb,
                                 const float* __restrict__ w1,
                                 const float* __restrict__ b1,
                                 const float* __restrict__ w2,
                                 float* __restrict__ table) {
    int r = blockIdx.x * blockDim.x + threadIdx.x;
    if (r >= 3375) return;
    float c0 = rpb[r * 3 + 0], c1 = rpb[r * 3 + 1], c2 = rpb[r * 3 + 2];
    float acc[HEADS] = {};
    for (int u = 0; u < 512; ++u) {
        float hsum = fmaxf(0.0f, c0 * w1[u * 3 + 0] + c1 * w1[u * 3 + 1] +
                                 c2 * w1[u * 3 + 2] + b1[u]);
#pragma unroll
        for (int h = 0; h < HEADS; ++h) acc[h] += hsum * w2[h * 512 + u];
    }
#pragma unroll
    for (int h = 0; h < HEADS; ++h) table[r * HEADS + h] = acc[h];
}

// biasH[h][q][k] = 16 * sigmoid(table[rpb_idx[q][k]][h]);  i = h<<18 | q<<9 | k
__global__ void bias_kernel(const float* __restrict__ table,
                            const int* __restrict__ rpb_idx,
                            float* __restrict__ biasH) {
    size_t i = (size_t)blockIdx.x * blockDim.x + threadIdx.x;
    int kq = (int)(i & (size_t)(NTOK * NTOK - 1));
    int h  = (int)(i >> 18);
    float t = table[(size_t)rpb_idx[kq] * HEADS + h];
    biasH[i] = 16.0f / (1.0f + __expf(-t));
}

// ---------------- QKV GEMM (bf16 WMMA) ----------------
// C[b][m][o] = sum_c xb[b][m][c] * qkv_w[o][c];  o -> (part, head, d)
// block: 256 thr = 8 waves, tile 128(M) x 128(N); wave = 32x64 = 2x4 WMMA tiles.
__global__ __launch_bounds__(256) void qkv_gemm_kernel(
    const bf16* __restrict__ xb, const bf16* __restrict__ wb,
    bf16* __restrict__ qb, bf16* __restrict__ kb, bf16* __restrict__ vtb) {
    const int b    = blockIdx.z;
    const int lane = threadIdx.x & 31;
    const int wave = threadIdx.x >> 5;
    const int half = lane >> 4;
    const int l16  = lane & 15;
    const int m0 = blockIdx.x * 128 + (wave >> 1) * 32;  // wave row base
    const int n0 = blockIdx.y * 128 + (wave & 1) * 64;   // wave col base

    const bf16* xrow = xb + (size_t)b * NTOK * DIM;
    v8f acc[2][4] = {};
    for (int kk = 0; kk < DIM; kk += 32) {
        v16bf a[2], bm[4];
#pragma unroll
        for (int i = 0; i < 2; ++i)
            a[i] = load_a_frag(xrow + (size_t)(m0 + i * 16 + l16) * DIM + kk, half);
#pragma unroll
        for (int j = 0; j < 4; ++j)
            bm[j] = load_b_frag(wb + (size_t)(n0 + j * 16 + l16) * DIM + kk, half);
#pragma unroll
        for (int i = 0; i < 2; ++i)
#pragma unroll
            for (int j = 0; j < 4; ++j)
                acc[i][j] = wmma_bf16(a[i], bm[j], acc[i][j]);
    }
    // scatter epilogue: C-layout element (row = m0+i*16+8*half+r, col = n0+j*16+l16)
#pragma unroll
    for (int i = 0; i < 2; ++i) {
        const int mrow = m0 + i * 16 + half * 8;
#pragma unroll
        for (int j = 0; j < 4; ++j) {
            const int o    = n0 + j * 16 + l16;
            const int part = o >> 8;           // wave-uniform
            const int h    = (o >> 5) & 7;
            const int d    = o & 31;
            const size_t bh = (size_t)b * HEADS + h;
#pragma unroll
            for (int r = 0; r < 8; ++r) {
                bf16 val = (bf16)acc[i][j][r];
                const int row = mrow + r;
                if (part == 0)       qb[(bh * NTOK + row) * HD + d] = val;
                else if (part == 1)  kb[(bh * NTOK + row) * HD + d] = val;
                else                 vtb[(bh * HD + d) * NTOK + row] = val;
            }
        }
    }
}

// ---------------- attention (flash-style, bf16 WMMA) ----------------
// grid: (N/128, B*H); block 256 thr = 8 waves; each wave owns 16 queries.
__global__ __launch_bounds__(256) void attn_kernel(
    const bf16* __restrict__ qb, const bf16* __restrict__ kb,
    const bf16* __restrict__ vtb, const float* __restrict__ biasH,
    const float* __restrict__ mask, bf16* __restrict__ ob) {
    const int bh = blockIdx.y;
    const int b  = bh >> 3, h = bh & 7;
    const int w  = b & (NW - 1);
    const int lane = threadIdx.x & 31;
    const int wave = threadIdx.x >> 5;
    const int half = lane >> 4;
    const int l16  = lane & 15;
    const int qbase = blockIdx.x * 128 + wave * 16;

    __shared__ __align__(32) bf16 pbuf[8][16 * 32];  // per-wave P tile (16q x 32k)

    const bf16* q  = qb  + (size_t)bh * NTOK * HD;
    const bf16* k  = kb  + (size_t)bh * NTOK * HD;
    const bf16* vt = vtb + (size_t)bh * HD * NTOK;
    const float* biasRow = biasH + (size_t)h * NTOK * NTOK;
    const float* maskRow = mask + (size_t)w * NTOK * NTOK;

    const v16bf qa = load_a_frag(q + (size_t)(qbase + l16) * HD, half);

    float m_i[8], l_i[8];
#pragma unroll
    for (int r = 0; r < 8; ++r) { m_i[r] = -1e30f; l_i[r] = 0.0f; }
    v8f o0 = {}, o1 = {};
    const float scale = 0.17677669529663687f;  // 1/sqrt(32)

    for (int jb = 0; jb < NTOK; jb += 32) {
        // scores: two 16x16 tiles over keys [jb, jb+32)
        v16bf kf0 = load_b_frag(k + (size_t)(jb + l16) * HD, half);
        v16bf kf1 = load_b_frag(k + (size_t)(jb + 16 + l16) * HD, half);
        v8f s0 = {}, s1 = {};
        s0 = wmma_bf16(qa, kf0, s0);
        s1 = wmma_bf16(qa, kf1, s1);

        const int qrow0 = qbase + 8 * half;
        float e0[8], e1[8];
#pragma unroll
        for (int r = 0; r < 8; ++r) {
            const int qr = qrow0 + r;
            const size_t rowoff = (size_t)qr * NTOK + jb + l16;
            float v0 = s0[r] * scale + biasRow[rowoff] + maskRow[rowoff];
            float v1 = s1[r] * scale + biasRow[rowoff + 16] + maskRow[rowoff + 16];
            float mx = fmaxf(v0, v1);
            for (int off = 1; off < 16; off <<= 1)
                mx = fmaxf(mx, __shfl_xor(mx, off, 32));
            mx = fmaxf(mx, m_i[r]);
            const float sc = __expf(m_i[r] - mx);
            const float p0 = __expf(v0 - mx);
            const float p1 = __expf(v1 - mx);
            float rs = p0 + p1;
            for (int off = 1; off < 16; off <<= 1) rs += __shfl_xor(rs, off, 32);
            l_i[r] = l_i[r] * sc + rs;
            m_i[r] = mx;
            o0[r] *= sc;
            o1[r] *= sc;
            e0[r] = p0;
            e1[r] = p1;
        }
        __syncthreads();
        // C-layout -> A-layout via LDS bounce (bf16)
#pragma unroll
        for (int r = 0; r < 8; ++r) {
            const int prow = 8 * half + r;
            pbuf[wave][prow * 32 + l16]      = (bf16)e0[r];
            pbuf[wave][prow * 32 + 16 + l16] = (bf16)e1[r];
        }
        __syncthreads();
        const v16bf pa = load_a_frag(&pbuf[wave][l16 * 32], half);
        // V B-fragments: vT[d][key] contiguous over keys
        const v16bf vb0 = load_b_frag(vt + (size_t)l16 * NTOK + jb, half);
        const v16bf vb1 = load_b_frag(vt + (size_t)(16 + l16) * NTOK + jb, half);
        o0 = wmma_bf16(pa, vb0, o0);
        o1 = wmma_bf16(pa, vb1, o1);
    }
    // normalize and store to (b, n, h*32+d) bf16
#pragma unroll
    for (int r = 0; r < 8; ++r) {
        const int qr = qbase + 8 * half + r;
        const float inv = 1.0f / l_i[r];
        const size_t base = ((size_t)b * NTOK + qr) * DIM + h * HD;
        ob[base + l16]      = (bf16)(o0[r] * inv);
        ob[base + 16 + l16] = (bf16)(o1[r] * inv);
    }
}

// ---------------- output projection (bf16 WMMA, fp32 out) ----------------
__global__ __launch_bounds__(256) void proj_gemm_kernel(
    const bf16* __restrict__ ob, const bf16* __restrict__ wb,
    const float* __restrict__ pb, float* __restrict__ out) {
    const int b    = blockIdx.z;
    const int lane = threadIdx.x & 31;
    const int wave = threadIdx.x >> 5;
    const int half = lane >> 4;
    const int l16  = lane & 15;
    const int m0 = blockIdx.x * 128 + (wave >> 1) * 32;
    const int n0 = blockIdx.y * 128 + (wave & 1) * 64;

    const bf16* orow = ob + (size_t)b * NTOK * DIM;
    v8f acc[2][4] = {};
    for (int kk = 0; kk < DIM; kk += 32) {
        v16bf a[2], bm[4];
#pragma unroll
        for (int i = 0; i < 2; ++i)
            a[i] = load_a_frag(orow + (size_t)(m0 + i * 16 + l16) * DIM + kk, half);
#pragma unroll
        for (int j = 0; j < 4; ++j)
            bm[j] = load_b_frag(wb + (size_t)(n0 + j * 16 + l16) * DIM + kk, half);
#pragma unroll
        for (int i = 0; i < 2; ++i)
#pragma unroll
            for (int j = 0; j < 4; ++j)
                acc[i][j] = wmma_bf16(a[i], bm[j], acc[i][j]);
    }
#pragma unroll
    for (int i = 0; i < 2; ++i) {
        const int mrow = m0 + i * 16 + half * 8;
#pragma unroll
        for (int j = 0; j < 4; ++j) {
            const int o = n0 + j * 16 + l16;
            const float bias = pb[o];
#pragma unroll
            for (int r = 0; r < 8; ++r)
                out[((size_t)b * NTOK + mrow + r) * DIM + o] = acc[i][j][r] + bias;
        }
    }
}

// ---------------- host-side launch ----------------
extern "C" void kernel_launch(void* const* d_in, const int* in_sizes, int n_in,
                              void* d_out, int out_size, void* d_ws, size_t ws_size,
                              hipStream_t stream) {
    (void)in_sizes; (void)n_in; (void)out_size; (void)ws_size;
    const float* x       = (const float*)d_in[0];
    const float* mask    = (const float*)d_in[1];
    const float* qkv_w   = (const float*)d_in[2];
    const float* proj_w  = (const float*)d_in[3];
    const float* proj_b  = (const float*)d_in[4];
    const float* cpb_w1  = (const float*)d_in[5];
    const float* cpb_b1  = (const float*)d_in[6];
    const float* cpb_w2  = (const float*)d_in[7];
    const float* rpb     = (const float*)d_in[8];
    const int*   rpb_idx = (const int*)d_in[9];
    float* out = (float*)d_out;

    char* ws = (char*)d_ws;
    float* table  = (float*)(ws + OFF_TABLE);
    float* biasH  = (float*)(ws + OFF_BIAS);
    bf16*  xb     = (bf16*)(ws + OFF_XB);
    bf16*  qkvwb  = (bf16*)(ws + OFF_QKVW);
    bf16*  projwb = (bf16*)(ws + OFF_PROJW);
    bf16*  qb     = (bf16*)(ws + OFF_Q);
    bf16*  kb     = (bf16*)(ws + OFF_K);
    bf16*  vtb    = (bf16*)(ws + OFF_VT);
    bf16*  obuf   = (bf16*)(ws + OFF_OB);

    // casts
    {
        int n = BATCH * NTOK * DIM;
        cast_f32_to_bf16<<<(n + 255) / 256, 256, 0, stream>>>(x, xb, n);
    }
    {
        int n = 3 * DIM * DIM;
        cast_f32_to_bf16<<<(n + 255) / 256, 256, 0, stream>>>(qkv_w, qkvwb, n);
    }
    {
        int n = DIM * DIM;
        cast_f32_to_bf16<<<(n + 255) / 256, 256, 0, stream>>>(proj_w, projwb, n);
    }
    // CPB table + bias tensor
    cpb_table_kernel<<<(3375 + 255) / 256, 256, 0, stream>>>(rpb, cpb_w1, cpb_b1,
                                                             cpb_w2, table);
    bias_kernel<<<(HEADS * NTOK * NTOK) / 256, 256, 0, stream>>>(table, rpb_idx,
                                                                 biasH);
    // QKV projection
    qkv_gemm_kernel<<<dim3(NTOK / 128, (3 * DIM) / 128, BATCH), 256, 0, stream>>>(
        xb, qkvwb, qb, kb, vtb);
    // attention
    attn_kernel<<<dim3(NTOK / 128, BATCH * HEADS), 256, 0, stream>>>(
        qb, kb, vtb, biasH, mask, obuf);
    // output projection
    proj_gemm_kernel<<<dim3(NTOK / 128, DIM / 128, BATCH), 256, 0, stream>>>(
        obuf, projwb, proj_b, out);
}
